// SNPAttention_16123307229953
// MI455X (gfx1250) — matrix operand order repaired and checked by reading, not compiled, not run on hardware
//
#include <hip/hip_runtime.h>

// SNP attention for MI455X (gfx1250, wave32, WMMA f16 16x16x32).
// B=8, H=8, N=128, L2=64, DK=64. Heads sequential via u-state in d_ws.

typedef _Float16 v16h __attribute__((ext_vector_type(16)));
typedef float    v8f  __attribute__((ext_vector_type(8)));

__device__ __forceinline__ v8f wmma32(v16h a, v16h b, v8f c) {
  // (neg_a, A, neg_b, B, c_mod, C, reuse_a, reuse_b)
  return __builtin_amdgcn_wmma_f32_16x16x32_f16(false, a, false, b, (short)0, c,
                                                false, false);
}

// A fragment: 16x32 (MxK), row-major source [64][64] f16.
// lane(0..15): M=row, K=0..15; lane(16..31): M=row, K=16..31
__device__ __forceinline__ v16h load_A(const _Float16* mat, int row, int kbase, bool hi) {
  const _Float16* p = mat + row * 64 + kbase + (hi ? 16 : 0);
  v16h a;
#pragma unroll
  for (int j = 0; j < 16; ++j) a[j] = p[j];
  return a;
}

// B fragment: 32x16 (KxN) where B = M^T and M is row-major [64][64] f16
// (i.e. B[k][n] = M[n][k]). lane(0..15): N=col, K=0..15; lane(16..31): K=16..31.
__device__ __forceinline__ v16h load_Bt(const _Float16* mat, int col, int kbase, bool hi) {
  const _Float16* p = mat + col * 64 + kbase + (hi ? 16 : 0);
  v16h b;
#pragma unroll
  for (int j = 0; j < 16; ++j) b[j] = p[j];
  return b;
}

// B fragment: 32x16 (KxN) from row-major [64][64] f16 (B[k][n] = M[k][n]).
__device__ __forceinline__ v16h load_Bv(const _Float16* mat, int col, int kbase, bool hi) {
  const int k0 = kbase + (hi ? 16 : 0);
  v16h b;
#pragma unroll
  for (int j = 0; j < 16; ++j) b[j] = mat[(k0 + j) * 64 + col];
  return b;
}

__global__ __launch_bounds__(256) void snp_init_kernel(float* u, float* u_acc) {
  const int tid = threadIdx.x;
  for (int i = tid; i < 4096; i += 256) { u[i] = 0.f; u_acc[i] = 0.f; }
}

__global__ __launch_bounds__(256) void snp_update_u_kernel(float* u, float* u_acc) {
  const int tid = threadIdx.x;
  for (int i = tid; i < 4096; i += 256) {
    u[i] = u_acc[i] * (1.0f / 1024.0f);  // mean over b*n = 8*128 rows
    u_acc[i] = 0.f;
  }
}

__global__ __launch_bounds__(256) void snp_head_kernel(
    const float* __restrict__ Q, const float* __restrict__ K,
    const float* __restrict__ V, float* __restrict__ Out,
    const float* __restrict__ u, float* __restrict__ u_acc, int head) {
  __shared__ float    xbuf[4096];   // x = u + S (f32)
  __shared__ float    ubuf[4096];   // u tile (f32)
  __shared__ _Float16 buf0[4096];   // Q(f16) then attn(f16)
  __shared__ _Float16 buf1[4096];   // K(f16) then V(f16)
  __shared__ float    red[512];

  const int tid  = threadIdx.x;
  const int ln   = tid & 31;
  const int w    = tid >> 5;
  const int lrow = ln & 15;
  const bool hi  = (ln & 16) != 0;

  const int bIdx = blockIdx.x >> 7;   // b in [0,8)
  const int nIdx = blockIdx.x & 127;  // n in [0,128)
  const size_t tile = ((size_t)(bIdx * 8 + head) * 128 + nIdx) * 4096;

  // ---- stage Q,K as f16 in LDS; u in LDS ----
  const float4* Q4 = (const float4*)(Q + tile);
  const float4* K4 = (const float4*)(K + tile);
  for (int i = tid; i < 1024; i += 256) {
    float4 q = Q4[i], k = K4[i];
    const int o = i * 4;
    buf0[o + 0] = (_Float16)q.x; buf0[o + 1] = (_Float16)q.y;
    buf0[o + 2] = (_Float16)q.z; buf0[o + 3] = (_Float16)q.w;
    buf1[o + 0] = (_Float16)k.x; buf1[o + 1] = (_Float16)k.y;
    buf1[o + 2] = (_Float16)k.z; buf1[o + 3] = (_Float16)k.w;
  }
  for (int i = tid; i < 4096; i += 256) ubuf[i] = u[i];
  __syncthreads();

  // ---- GEMM1: S = Q * K^T  (each wave: 16x32 slab = 2 tiles) ----
  const int mt = w >> 1;            // 0..3
  const int nt = (w & 1) * 2;       // 0 or 2
  v8f c0 = {}; v8f c1 = {};
#pragma unroll
  for (int ks = 0; ks < 64; ks += 32) {
    v16h a  = load_A (buf0, mt * 16 + lrow, ks, hi);
    v16h b0 = load_Bt(buf1, nt * 16 + lrow, ks, hi);
    v16h b1 = load_Bt(buf1, (nt + 1) * 16 + lrow, ks, hi);
    c0 = wmma32(a, b0, c0);
    c1 = wmma32(a, b1, c1);
  }

  // x = S/8 + u  -> xbuf
#pragma unroll
  for (int r = 0; r < 8; ++r) {
    const int qrow = mt * 16 + r + (hi ? 8 : 0);
    const int p0 = qrow * 64 + nt * 16 + lrow;
    const int p1 = p0 + 16;
    xbuf[p0] = c0[r] * 0.125f + ubuf[p0];
    xbuf[p1] = c1[r] * 0.125f + ubuf[p1];
  }
  __syncthreads();

  // ---- two flattened-4096 softmax reductions: y1 = x, y2 = x + u ----
  float lm1 = -1e30f, lm2 = -1e30f;
  for (int i = tid; i < 4096; i += 256) {
    const float x = xbuf[i];
    const float y = x + ubuf[i];
    lm1 = fmaxf(lm1, x);
    lm2 = fmaxf(lm2, y);
  }
  red[tid] = lm1; red[tid + 256] = lm2;
  __syncthreads();
  for (int s = 128; s > 0; s >>= 1) {
    if (tid < s) {
      red[tid]       = fmaxf(red[tid],       red[tid + s]);
      red[tid + 256] = fmaxf(red[tid + 256], red[tid + 256 + s]);
    }
    __syncthreads();
  }
  const float m1 = red[0], m2 = red[256];
  __syncthreads();

  float ls1 = 0.f, ls2 = 0.f;
  for (int i = tid; i < 4096; i += 256) {
    const float x = xbuf[i];
    ls1 += __expf(x - m1);
    ls2 += __expf(x + ubuf[i] - m2);
  }
  red[tid] = ls1; red[tid + 256] = ls2;
  __syncthreads();
  for (int s = 128; s > 0; s >>= 1) {
    if (tid < s) {
      red[tid]       += red[tid + s];
      red[tid + 256] += red[tid + 256 + s];
    }
    __syncthreads();
  }
  const float inv1 = 1.0f / red[0];
  const float inv2 = 1.0f / red[256];
  __syncthreads();

  // ---- u_acc += (x - softmax(x)); attn = softmax(x+u) -> buf0; V -> buf1 ----
  for (int i = tid; i < 4096; i += 256) {
    const float x  = xbuf[i];
    const float p  = __expf(x - m1) * inv1;
    unsafeAtomicAdd(&u_acc[i], x - p);               // global_atomic_add_f32
    const float at = __expf(x + ubuf[i] - m2) * inv2;
    buf0[i] = (_Float16)at;
  }
  const float4* V4 = (const float4*)(V + tile);
  for (int i = tid; i < 1024; i += 256) {
    float4 v = V4[i];
    const int o = i * 4;
    buf1[o + 0] = (_Float16)v.x; buf1[o + 1] = (_Float16)v.y;
    buf1[o + 2] = (_Float16)v.z; buf1[o + 3] = (_Float16)v.w;
  }
  __syncthreads();

  // ---- GEMM2: context = attn * V ----
  v8f d0 = {}; v8f d1 = {};
#pragma unroll
  for (int ks = 0; ks < 64; ks += 32) {
    v16h a  = load_A (buf0, mt * 16 + lrow, ks, hi);
    v16h b0 = load_Bv(buf1, nt * 16 + lrow, ks, hi);
    v16h b1 = load_Bv(buf1, (nt + 1) * 16 + lrow, ks, hi);
    d0 = wmma32(a, b0, d0);
    d1 = wmma32(a, b1, d1);
  }

  // stage into xbuf for coalesced f32x4 store
#pragma unroll
  for (int r = 0; r < 8; ++r) {
    const int qrow = mt * 16 + r + (hi ? 8 : 0);
    const int p0 = qrow * 64 + nt * 16 + lrow;
    xbuf[p0]      = d0[r];
    xbuf[p0 + 16] = d1[r];
  }
  __syncthreads();
  float4* O4 = (float4*)(Out + tile);
  const float4* X4 = (const float4*)xbuf;
  for (int i = tid; i < 1024; i += 256) O4[i] = X4[i];
}

extern "C" void kernel_launch(void* const* d_in, const int* in_sizes, int n_in,
                              void* d_out, int out_size, void* d_ws, size_t ws_size,
                              hipStream_t stream) {
  const float* Q = (const float*)d_in[0];
  const float* K = (const float*)d_in[1];
  const float* V = (const float*)d_in[2];
  float* Out = (float*)d_out;

  float* u     = (float*)d_ws;        // 4096 f32
  float* u_acc = u + 4096;            // 4096 f32

  snp_init_kernel<<<1, 256, 0, stream>>>(u, u_acc);
  for (int h = 0; h < 8; ++h) {
    snp_head_kernel<<<1024, 256, 0, stream>>>(Q, K, V, Out, u, u_acc, h);
    snp_update_u_kernel<<<1, 256, 0, stream>>>(u, u_acc);
  }
}
